// MAGNN_nc_mb_22187801051275
// MI455X (gfx1250) — matrix operand order repaired and checked by reading, not compiled, not run on hardware
//
#include <hip/hip_runtime.h>

// ---------------------------------------------------------------------------
// MAGNN_nc_mb forward for MI455X (gfx1250), wave32 + WMMA bf16
// Round 3: GRU kernel stages A tiles in LDS (shared by all waves, bank-conflict
// free ds_load_b128) and computes 16x32 per wave (2x arithmetic intensity).
// ---------------------------------------------------------------------------

typedef __bf16 bf16_t;
typedef __attribute__((ext_vector_type(16))) __bf16 bf16x16;
typedef __attribute__((ext_vector_type(8)))  __bf16 bf16x8;
typedef __attribute__((ext_vector_type(8)))  float  f32x8;

#define N0C 2500
#define N1C 3500
#define N2C 3000
#define NTOT 9000
#define NHD 8
#define PMP 2
#define LMP 3
#define ECNT 50000
#define NLC 5000
#define BTGT 1024
#define G3H 1536       // 3*H
#define FRAGE 512      // elems per packed fragment (32 lanes * 16)
#define MPAD 2560      // attention src rows padded (80 tiles of 32)
#define XLS 72         // LDS stride for X tile (bank-spread, 16B aligned)
#define HLS 520        // LDS stride for H tile (bank-spread, 16B aligned)

// ------------------------------ WMMA helpers -------------------------------

__device__ __forceinline__ f32x8 wmma_bf16(bf16x16 a, bf16x16 b, f32x8 c) {
  return __builtin_amdgcn_wmma_f32_16x16x32_bf16(false, a, false, b, (short)0, c,
                                                 false, false);
}

// Packed B fragment: one aligned 32-byte vector load per lane.
__device__ __forceinline__ bf16x16 ldB_packed(const bf16_t* __restrict__ P, int frag) {
  int lane = threadIdx.x & 31;
  return *(const bf16x16*)(P + ((long)frag * 32 + (long)lane) * 16);
}

// A fragment 16x32 bf16 (row-major, stride elems): two 16B vector loads.
__device__ __forceinline__ bf16x16 ldA_bf16v(const bf16_t* __restrict__ A, int stride,
                                             int row0, int k0, int maxrow) {
  int lane = threadIdx.x & 31;
  int m = row0 + (lane & 15);
  if (m > maxrow) m = maxrow;
  int hh = lane >> 4;
  const bf16_t* p = A + (long)m * stride + k0 + 16 * hh;
  bf16x8 lo = *(const bf16x8*)(p);
  bf16x8 hi = *(const bf16x8*)(p + 8);
  bf16x16 v;
#pragma unroll
  for (int j = 0; j < 8; ++j) { v[j] = lo[j]; v[8 + j] = hi[j]; }
  return v;
}

// A fragment from f32 source, full 32-wide K chunk in-bounds: 4x float4 loads.
__device__ __forceinline__ bf16x16 ldA_f32v(const float* __restrict__ A, int stride,
                                            int row0, int k0, int maxrow) {
  int lane = threadIdx.x & 31;
  int m = row0 + (lane & 15);
  if (m > maxrow) m = maxrow;
  int hh = lane >> 4;
  const float* p = A + (long)m * stride + k0 + 16 * hh;
  float4 a = *(const float4*)(p);
  float4 b = *(const float4*)(p + 4);
  float4 c = *(const float4*)(p + 8);
  float4 d = *(const float4*)(p + 12);
  bf16x16 v;
  v[0] = (bf16_t)a.x;  v[1] = (bf16_t)a.y;  v[2] = (bf16_t)a.z;  v[3] = (bf16_t)a.w;
  v[4] = (bf16_t)b.x;  v[5] = (bf16_t)b.y;  v[6] = (bf16_t)b.z;  v[7] = (bf16_t)b.w;
  v[8] = (bf16_t)c.x;  v[9] = (bf16_t)c.y;  v[10] = (bf16_t)c.z; v[11] = (bf16_t)c.w;
  v[12] = (bf16_t)d.x; v[13] = (bf16_t)d.y; v[14] = (bf16_t)d.z; v[15] = (bf16_t)d.w;
  return v;
}

// K-tail A fragment: branch-free (clamped index + value select), used once.
__device__ __forceinline__ bf16x16 ldA_f32_tail(const float* __restrict__ A, int stride,
                                                int row0, int k0, int maxrow, int K) {
  int lane = threadIdx.x & 31;
  int m = row0 + (lane & 15);
  if (m > maxrow) m = maxrow;
  int hh = lane >> 4;
  const float* p = A + (long)m * stride;
  bf16x16 v;
#pragma unroll
  for (int j = 0; j < 8; ++j) {
    int k = k0 + 2 * j + 16 * hh;
    float x0 = p[(k < K) ? k : (K - 1)];
    float x1 = p[(k + 1 < K) ? (k + 1) : (K - 1)];
    v[2 * j]     = (bf16_t)((k < K) ? x0 : 0.f);
    v[2 * j + 1] = (bf16_t)((k + 1 < K) ? x1 : 0.f);
  }
  return v;
}

// B[k][n] = Km[n0+n][k] (K-matrix transposed on the fly): one 32B vector load.
__device__ __forceinline__ bf16x16 ldBT_bf16v(const bf16_t* __restrict__ Km, int stride,
                                              int n0, int k0, int maxrow) {
  int lane = threadIdx.x & 31;
  int n = n0 + (lane & 15);
  if (n > maxrow) n = maxrow;
  int hh = lane >> 4;
  return *(const bf16x16*)(Km + (long)n * stride + k0 + hh * 16);
}

__device__ __forceinline__ float sigf(float x) { return 1.f / (1.f + __expf(-x)); }

__device__ __forceinline__ unsigned f2ord(float f) {
  unsigned u = __float_as_uint(f);
  return (u & 0x80000000u) ? ~u : (u | 0x80000000u);
}
__device__ __forceinline__ float ord2f(unsigned u) {
  return (u & 0x80000000u) ? __uint_as_float(u & 0x7fffffffu)
                           : __uint_as_float(~u);
}

// --------------------------- weight pack kernels ---------------------------
__global__ void k_packB(const float* __restrict__ B, int K, int N, int KT, int nb,
                        bf16_t* __restrict__ out) {
  long t = (long)blockIdx.x * blockDim.x + threadIdx.x;
  int ntil = N >> 4;
  long fragsPer = (long)KT * ntil;
  if (t >= fragsPer * nb * 32) return;
  int lane = (int)(t & 31);
  long frag = t >> 5;
  int b = (int)(frag / fragsPer);
  int fr = (int)(frag % fragsPer);
  int nt = fr % ntil, kt = fr / ntil;
  int n = nt * 16 + (lane & 15);
  int ks = kt * 32 + (lane >> 4) * 16;
  const float* Bb = B + (long)b * K * N;
  bf16_t* dst = out + t * 16;
#pragma unroll 4
  for (int j = 0; j < 16; ++j) {
    int k = ks + j;
    dst[j] = (k < K) ? (bf16_t)Bb[(long)k * N + n] : (bf16_t)0.f;
  }
}

__global__ void k_packBT(const float* __restrict__ Wt, int K, int N,
                         bf16_t* __restrict__ out) {
  long t = (long)blockIdx.x * blockDim.x + threadIdx.x;
  int ntil = N >> 4;
  long frags = (long)(K >> 5) * ntil;
  if (t >= frags * 32) return;
  int lane = (int)(t & 31);
  long frag = t >> 5;
  int nt = (int)(frag % ntil), kt = (int)(frag / ntil);
  int n = nt * 16 + (lane & 15);
  int ks = kt * 32 + (lane >> 4) * 16;
  bf16_t* dst = out + t * 16;
#pragma unroll 4
  for (int j = 0; j < 16; ++j) dst[j] = (bf16_t)Wt[(long)n * K + ks + j];
}

__global__ void k_pack_attnW(const float* __restrict__ attn, bf16_t* __restrict__ out) {
  int t = blockIdx.x * blockDim.x + threadIdx.x;  // 16 frags * 32 lanes
  if (t >= 16 * 32) return;
  int lane = t & 31, kt = t >> 5;
  int n = lane & 15;
  int ks = kt * 32 + (lane >> 4) * 16;
  bf16_t* dst = out + (long)t * 16;
#pragma unroll 4
  for (int j = 0; j < 16; ++j) {
    int k = ks + j;
    float v = 0.f;
    if (n < NHD && (k >> 6) == n) v = attn[n * 64 + (k & 63)];
    dst[j] = (bf16_t)v;
  }
}

__global__ void k_packV(const bf16_t* __restrict__ V, bf16_t* __restrict__ out) {
  long t = (long)blockIdx.x * blockDim.x + threadIdx.x;  // 320 frags * 32
  if (t >= (long)(MPAD / 32) * 4 * 32) return;
  int lane = (int)(t & 31);
  long frag = t >> 5;
  int nt = (int)(frag & 3), kt = (int)(frag >> 2);
  int n = nt * 16 + (lane & 15);
  int ks = kt * 32 + (lane >> 4) * 16;
  bf16_t* dst = out + t * 16;
#pragma unroll 4
  for (int j = 0; j < 16; ++j) {
    int m = ks + j;
    dst[j] = (m < N0C) ? V[(long)m * 64 + n] : (bf16_t)0.f;
  }
}

// ------------------------- 1) per-type feature GEMM ------------------------
__global__ void k_feat(const float* __restrict__ F, const bf16_t* __restrict__ Wp,
                       const float* __restrict__ bias, int M, int K, int row_off,
                       float* __restrict__ tf, bf16_t* __restrict__ tf16) {
  int wid = threadIdx.x >> 5;
  int rt = blockIdx.x * 64 + wid * 16;
  if (rt >= M) return;
  f32x8 acc[4] = {};
  int ktmain = K >> 5;
  int KT = (K + 31) >> 5;
  for (int kt = 0; kt < ktmain; ++kt) {
    bf16x16 a = ldA_f32v(F, K, rt, kt * 32, M - 1);
#pragma unroll
    for (int nt = 0; nt < 4; ++nt)
      acc[nt] = wmma_bf16(a, ldB_packed(Wp, kt * 4 + nt), acc[nt]);
  }
  if (ktmain < KT) {  // zero-padded packed B kills out-of-range products
    bf16x16 a = ldA_f32_tail(F, K, rt, ktmain * 32, M - 1, K);
#pragma unroll
    for (int nt = 0; nt < 4; ++nt)
      acc[nt] = wmma_bf16(a, ldB_packed(Wp, ktmain * 4 + nt), acc[nt]);
  }
  int lane = threadIdx.x & 31;
#pragma unroll
  for (int nt = 0; nt < 4; ++nt)
#pragma unroll
    for (int r = 0; r < 8; ++r) {
      int m = rt + r + 8 * (lane >> 4);
      int c = nt * 16 + (lane & 15);
      if (m < M) {
        float v = acc[nt][r] + bias[c];
        tf[(long)(row_off + m) * 64 + c] = v;
        tf16[(long)(row_off + m) * 64 + c] = (bf16_t)v;
      }
    }
}

// -------------------- 2) q0 = emb @ acW[h] ---------------------------------
__global__ void k_q0(const float* __restrict__ emb, const bf16_t* __restrict__ acWp,
                     bf16_t* __restrict__ q0) {
  int h = blockIdx.y;
  int wid = threadIdx.x >> 5;
  int rt = blockIdx.x * 64 + wid * 16;
  if (rt >= NTOT) return;
  const bf16_t* Ph = acWp + (long)h * 16 * FRAGE;
  f32x8 acc[8] = {};
#pragma unroll
  for (int kt = 0; kt < 2; ++kt) {
    bf16x16 a = ldA_f32v(emb, 64, rt, kt * 32, NTOT - 1);
#pragma unroll
    for (int nt = 0; nt < 8; ++nt)
      acc[nt] = wmma_bf16(a, ldB_packed(Ph, kt * 8 + nt), acc[nt]);
  }
  int lane = threadIdx.x & 31;
  bf16_t* out = q0 + (long)h * NTOT * 128;
#pragma unroll
  for (int nt = 0; nt < 8; ++nt)
#pragma unroll
    for (int r = 0; r < 8; ++r) {
      int m = rt + r + 8 * (lane >> 4);
      int c = nt * 16 + (lane & 15);
      if (m < NTOT) out[(long)m * 128 + c] = (bf16_t)acc[nt][r];
    }
}

// -------------------- 3) q = q0 @ acW2[h] ----------------------------------
__global__ void k_q(const bf16_t* __restrict__ q0, const bf16_t* __restrict__ acW2p,
                    bf16_t* __restrict__ q) {
  int h = blockIdx.y;
  int wid = threadIdx.x >> 5;
  int rt = blockIdx.x * 64 + wid * 16;
  if (rt >= NTOT) return;
  const bf16_t* A = q0 + (long)h * NTOT * 128;
  const bf16_t* Ph = acW2p + (long)h * 32 * FRAGE;
  f32x8 acc[8] = {};
#pragma unroll
  for (int kt = 0; kt < 4; ++kt) {
    bf16x16 a = ldA_bf16v(A, 128, rt, kt * 32, NTOT - 1);
#pragma unroll
    for (int nt = 0; nt < 8; ++nt)
      acc[nt] = wmma_bf16(a, ldB_packed(Ph, kt * 8 + nt), acc[nt]);
  }
  int lane = threadIdx.x & 31;
  bf16_t* out = q + (long)h * NTOT * 128;
#pragma unroll
  for (int nt = 0; nt < 8; ++nt)
#pragma unroll
    for (int r = 0; r < 8; ++r) {
      int m = rt + r + 8 * (lane >> 4);
      int c = nt * 16 + (lane & 15);
      if (m < NTOT) out[(long)m * 128 + c] = (bf16_t)acc[nt][r];
    }
}

// ------------- 4) fused attention: fsr += softmax(sig(qk^T)) @ V / NH ------
__global__ void k_attn(const bf16_t* __restrict__ q, const bf16_t* __restrict__ kmat,
                       const bf16_t* __restrict__ Vp, float* __restrict__ fsr) {
  int h = blockIdx.y;
  int n0 = blockIdx.x * 16;
  int wid = threadIdx.x >> 5;
  int lane = threadIdx.x & 31;
  __shared__ bf16_t Pl[4][16][40];   // 80B rows -> 16B-aligned chunks
  __shared__ float denom[16];
  if (threadIdx.x < 16) denom[threadIdx.x] = 0.f;
  __syncthreads();
  const bf16_t* qh = q + (long)h * NTOT * 128;
  const bf16_t* kh = kmat + (long)h * NTOT * 128;
  bf16x16 qf[4];
#pragma unroll
  for (int kk = 0; kk < 4; ++kk) qf[kk] = ldA_bf16v(qh, 128, n0, kk * 32, NTOT - 1);
  f32x8 oacc[4] = {};
  for (int mb = 0; mb < MPAD; mb += 128) {  // covers 2500 src rows
    int mybase = mb + 32 * wid;
#pragma unroll
    for (int s = 0; s < 2; ++s) {
      int mt = mybase + 16 * s;
      f32x8 sc = {};
#pragma unroll
      for (int kk = 0; kk < 4; ++kk)
        sc = wmma_bf16(qf[kk], ldBT_bf16v(kh, 128, mt, kk * 32, N0C - 1), sc);
#pragma unroll
      for (int r = 0; r < 8; ++r) {
        int mm = mt + (lane & 15);
        int row = r + 8 * (lane >> 4);
        float p = 0.f;
        if (mm < N0C) p = __expf(sigf(sc[r]));  // sigmoid in (0,1): no max sub
        Pl[wid][row][16 * s + (lane & 15)] = (bf16_t)p;
        atomicAdd(&denom[row], p);
      }
    }
    __syncthreads();
    bf16x16 pf = ldA_bf16v(&Pl[wid][0][0], 40, 0, 0, 15);
    int vf = (mybase >> 5) * 4;
#pragma unroll
    for (int nt = 0; nt < 4; ++nt)
      oacc[nt] = wmma_bf16(pf, ldB_packed(Vp, vf + nt), oacc[nt]);
    __syncthreads();
  }
#pragma unroll
  for (int nt = 0; nt < 4; ++nt)
#pragma unroll
    for (int r = 0; r < 8; ++r) {
      int row = r + 8 * (lane >> 4);
      int n = n0 + row;
      int c = nt * 16 + (lane & 15);
      if (n < NTOT)
        atomicAdd(&fsr[(long)n * 64 + c],
                  oacc[nt][r] / ((denom[row] + 1e-9f) * (float)NHD));
    }
}

// ---------------- 5) tf[N0:] += fsr[N0:]; refresh bf16 copy ----------------
__global__ void k_addfsr(float* __restrict__ tf, bf16_t* __restrict__ tf16,
                         const float* __restrict__ fsr) {
  long i = (long)blockIdx.x * blockDim.x + threadIdx.x;
  const long total = (long)(NTOT - N0C) * 64;
  if (i >= total) return;
  long off = (long)N0C * 64 + i;
  float v = tf[off] + fsr[off];
  tf[off] = v;
  tf16[off] = (bf16_t)v;
}

// ---------------- 6) gather metapath instance features (16B/thread) --------
__global__ void k_gather(const int* __restrict__ emi, const bf16_t* __restrict__ tf16,
                         bf16_t* __restrict__ edata) {
  long i = (long)blockIdx.x * blockDim.x + threadIdx.x;  // (e*3+t)*8 + chunk
  if (i >= (long)ECNT * LMP * 8) return;
  int c = (int)(i & 7);
  long et = i >> 3;
  int node = emi[et];
  *(bf16x8*)(edata + et * 64 + c * 8) =
      *(const bf16x8*)(tf16 + (long)node * 64 + c * 8);
}

// ---------------- 7) GRU step (torch gate order r,z,n) ---------------------
// Block: 16 E-rows x 128 hidden cols. A tiles staged once in LDS (shared by
// all 4 waves); each wave computes a 16x32 output slab (2 column tiles).
__global__ void k_gru(const bf16_t* __restrict__ X, int xstride,
                      const bf16_t* __restrict__ Hin,
                      const bf16_t* __restrict__ wihP, const bf16_t* __restrict__ whhP,
                      const float* __restrict__ bih, const float* __restrict__ bhh,
                      bf16_t* __restrict__ Hout, int first) {
  int tid = threadIdx.x;
  int wid = tid >> 5;
  int lane = tid & 31;
  int ct0 = (blockIdx.x * 8 + wid * 2) * 16;  // wave covers cols [ct0, ct0+32)
  int nt0 = ct0 >> 4;
  int rt = blockIdx.y * 16;                   // E row tile
  __shared__ bf16_t Xl[16][XLS];
  __shared__ bf16_t Hl[16][HLS];
  {  // stage X tile (16x64): 128 chunks of 16B
    int row = tid >> 3, c8 = (tid & 7) * 8;
    *(bf16x8*)(&Xl[row][c8]) = *(const bf16x8*)(X + (long)(rt + row) * xstride + c8);
  }
  if (!first) {  // stage H rows 0-7 (8x512): 512 chunks of 16B
#pragma unroll
    for (int it = 0; it < 4; ++it) {
      int c = tid + it * 128;
      int row = c >> 6, c8 = (c & 63) * 8;
      *(bf16x8*)(&Hl[row][c8]) = *(const bf16x8*)(Hin + (long)(rt + row) * 512 + c8);
    }
  }
  __syncthreads();
  f32x8 gi[6] = {};  // [tile*3 + gate(r,z,n)]
  f32x8 gh[6] = {};
#pragma unroll
  for (int kt = 0; kt < 2; ++kt) {
    bf16x16 a = ldA_bf16v(&Xl[0][0], XLS, 0, kt * 32, 15);
#pragma unroll
    for (int tl = 0; tl < 2; ++tl) {
      int fb = kt * 96 + nt0 + tl;
      gi[tl * 3 + 0] = wmma_bf16(a, ldB_packed(wihP, fb), gi[tl * 3 + 0]);
      gi[tl * 3 + 1] = wmma_bf16(a, ldB_packed(wihP, fb + 32), gi[tl * 3 + 1]);
      gi[tl * 3 + 2] = wmma_bf16(a, ldB_packed(wihP, fb + 64), gi[tl * 3 + 2]);
    }
  }
  if (!first) {
    for (int kt = 0; kt < 16; ++kt) {
      bf16x16 a = ldA_bf16v(&Hl[0][0], HLS, 0, kt * 32, 15);
#pragma unroll
      for (int tl = 0; tl < 2; ++tl) {
        int fb = kt * 96 + nt0 + tl;
        gh[tl * 3 + 0] = wmma_bf16(a, ldB_packed(whhP, fb), gh[tl * 3 + 0]);
        gh[tl * 3 + 1] = wmma_bf16(a, ldB_packed(whhP, fb + 32), gh[tl * 3 + 1]);
        gh[tl * 3 + 2] = wmma_bf16(a, ldB_packed(whhP, fb + 64), gh[tl * 3 + 2]);
      }
    }
  }
#pragma unroll
  for (int tl = 0; tl < 2; ++tl)
#pragma unroll
    for (int r = 0; r < 8; ++r) {
      int row = r + 8 * (lane >> 4);
      int m = rt + row;
      int c = ct0 + tl * 16 + (lane & 15);
      float rr = sigf(gi[tl * 3 + 0][r] + bih[c] + gh[tl * 3 + 0][r] + bhh[c]);
      float zz = sigf(gi[tl * 3 + 1][r] + bih[512 + c] + gh[tl * 3 + 1][r] + bhh[512 + c]);
      float gg = tanhf(gi[tl * 3 + 2][r] + bih[1024 + c] +
                       rr * (gh[tl * 3 + 2][r] + bhh[1024 + c]));
      float hold = first ? 0.f : (float)Hl[row][c];
      Hout[(long)m * 512 + c] = (bf16_t)((1.f - zz) * gg + zz * hold);
    }
}

// ------------- 8) edge attention scores via WMMA + segment max -------------
__global__ void k_scoresW(const bf16_t* __restrict__ eft,
                          const bf16_t* __restrict__ attnWp,
                          const int* __restrict__ edst, float* __restrict__ a,
                          unsigned* __restrict__ amax) {
  int wid = threadIdx.x >> 5;
  int rt = (blockIdx.x * 4 + wid) * 16;
  if (rt >= ECNT) return;
  f32x8 acc = {};
  for (int kt = 0; kt < 16; ++kt)
    acc = wmma_bf16(ldA_bf16v(eft, 512, rt, kt * 32, ECNT - 1),
                    ldB_packed(attnWp, kt), acc);
  int lane = threadIdx.x & 31;
  int h = lane & 15;
#pragma unroll
  for (int r = 0; r < 8; ++r) {
    int e = rt + r + 8 * (lane >> 4);
    if (h < NHD && e < ECNT) {
      float s = acc[r];
      float lr = s > 0.f ? s : 0.01f * s;  // leaky_relu
      a[(long)e * 8 + h] = lr;
      atomicMax(&amax[(long)edst[e] * 8 + h], f2ord(lr));
    }
  }
}

// ------------- 9) exp + segment sum (softmax denominator) ------------------
__global__ void k_expden(const float* __restrict__ a, const unsigned* __restrict__ amax,
                         const int* __restrict__ edst, float* __restrict__ ea,
                         float* __restrict__ den) {
  long i = (long)blockIdx.x * blockDim.x + threadIdx.x;
  if (i >= (long)ECNT * NHD) return;
  int h = (int)(i & 7);
  long e = i >> 3;
  int dst = edst[e];
  float v = __expf(a[i] - ord2f(amax[(long)dst * 8 + h]));
  ea[i] = v;
  atomicAdd(&den[(long)dst * 8 + h], v);
}

// ------------- 10) weighted scatter-aggregate ------------------------------
__global__ void k_agg(const bf16_t* __restrict__ eft, const float* __restrict__ ea,
                      const float* __restrict__ den, const int* __restrict__ edst,
                      float* __restrict__ ft) {
  long i = (long)blockIdx.x * blockDim.x + threadIdx.x;  // e*512+c
  if (i >= (long)ECNT * 512) return;
  int c = (int)(i & 511);
  long e = i >> 9;
  int h = c >> 6;
  int dst = edst[e];
  float w = ea[e * 8 + h] / (den[(long)dst * 8 + h] + 1e-9f);
  atomicAdd(&ft[(long)dst * 512 + c], (float)eft[i] * w);
}

// ------------- 11) target gather + ELU -------------------------------------
__global__ void k_gelu(const float* __restrict__ ft, const int* __restrict__ tidx,
                       float* __restrict__ m) {
  long i = (long)blockIdx.x * blockDim.x + threadIdx.x;
  if (i >= (long)BTGT * 512) return;
  int c = (int)(i & 511);
  int b = (int)(i >> 9);
  float v = ft[(long)tidx[b] * 512 + c];
  m[i] = v > 0.f ? v : (__expf(v) - 1.f);
}

// ------------- 12) semantic attention score s_p ----------------------------
__global__ void k_sem(const float* __restrict__ m, const float* __restrict__ fc1W,
                      const float* __restrict__ fc1b, const float* __restrict__ fc2w,
                      float* __restrict__ sout) {
  int b = blockIdx.x;
  int c = threadIdx.x;  // 0..127
  const float* row = m + (long)b * 512;
  float acc = fc1b[c];
  for (int k = 0; k < 512; ++k) acc += row[k] * fc1W[k * 128 + c];
  float val = tanhf(acc) * fc2w[c];
  __shared__ float red[128];
  red[c] = val;
  __syncthreads();
  for (int s = 64; s > 0; s >>= 1) {
    if (c < s) red[c] += red[c + s];
    __syncthreads();
  }
  if (c == 0) atomicAdd(sout, red[0] * (1.f / (float)BTGT));
}

__global__ void k_beta(const float* __restrict__ s, float* __restrict__ beta) {
  if (threadIdx.x == 0) {
    float mx = fmaxf(s[0], s[1]);
    float e0 = __expf(s[0] - mx), e1 = __expf(s[1] - mx);
    beta[0] = e0 / (e0 + e1);
    beta[1] = e1 / (e0 + e1);
  }
}

// ------------- 13) hagg + logits -------------------------------------------
__global__ void k_final(const float* __restrict__ m0, const float* __restrict__ m1,
                        const float* __restrict__ beta, const float* __restrict__ outW,
                        const float* __restrict__ outb, float* __restrict__ out) {
  int b = blockIdx.x;
  int t = threadIdx.x;  // 128 threads
  __shared__ float hrow[512];
  float b0 = beta[0], b1 = beta[1];
  for (int c = t; c < 512; c += 128) {
    float v = b0 * m0[(long)b * 512 + c] + b1 * m1[(long)b * 512 + c];
    hrow[c] = v;
    out[(long)BTGT * 3 + (long)b * 512 + c] = v;
  }
  __syncthreads();
  __shared__ float red[3][128];
  float p0 = 0.f, p1 = 0.f, p2 = 0.f;
  for (int c = t; c < 512; c += 128) {
    float v = hrow[c];
    p0 += v * outW[c * 3 + 0];
    p1 += v * outW[c * 3 + 1];
    p2 += v * outW[c * 3 + 2];
  }
  red[0][t] = p0; red[1][t] = p1; red[2][t] = p2;
  __syncthreads();
  for (int s = 64; s > 0; s >>= 1) {
    if (t < s) {
      red[0][t] += red[0][t + s];
      red[1][t] += red[1][t + s];
      red[2][t] += red[2][t + s];
    }
    __syncthreads();
  }
  if (t < 3) out[(long)b * 3 + t] = red[t][0] + outb[t];
}

// ---------------------------------------------------------------------------

extern "C" void kernel_launch(void* const* d_in, const int* in_sizes, int n_in,
                              void* d_out, int out_size, void* d_ws, size_t ws_size,
                              hipStream_t stream) {
  (void)in_sizes; (void)n_in; (void)out_size; (void)ws_size;
  const float* f0     = (const float*)d_in[0];
  const float* f1     = (const float*)d_in[1];
  const float* f2     = (const float*)d_in[2];
  const float* emb    = (const float*)d_in[3];
  const int*   emi[2] = {(const int*)d_in[4], (const int*)d_in[5]};
  const int*   edst[2]= {(const int*)d_in[6], (const int*)d_in[7]};
  const int*   tidx[2]= {(const int*)d_in[8], (const int*)d_in[9]};
  const float* fcW0   = (const float*)d_in[10];
  const float* fcb0   = (const float*)d_in[11];
  const float* fcW1   = (const float*)d_in[12];
  const float* fcb1   = (const float*)d_in[13];
  const float* fcW2   = (const float*)d_in[14];
  const float* fcb2   = (const float*)d_in[15];
  const float* acW    = (const float*)d_in[16];
  const float* acW2   = (const float*)d_in[17];
  const float* gwih   = (const float*)d_in[18];
  const float* gwhh   = (const float*)d_in[19];
  const float* gbih   = (const float*)d_in[20];
  const float* gbhh   = (const float*)d_in[21];
  const float* attnp  = (const float*)d_in[22];
  const float* fc1W   = (const float*)d_in[23];
  const float* fc1b   = (const float*)d_in[24];
  const float* fc2w   = (const float*)d_in[25];
  const float* outW   = (const float*)d_in[26];
  const float* outb   = (const float*)d_in[27];
  float* out = (float*)d_out;

  // workspace bump allocator
  char* base = (char*)d_ws;
  size_t off = 0;
  auto alloc = [&](size_t bytes) -> void* {
    void* r = base + off;
    off += (bytes + 255) & ~(size_t)255;
    return r;
  };
  float*  tf    = (float*)alloc((size_t)NTOT * 64 * 4);
  bf16_t* tf16  = (bf16_t*)alloc((size_t)NTOT * 64 * 2);
  bf16_t* q0    = (bf16_t*)alloc((size_t)NHD * NTOT * 128 * 2);
  bf16_t* qb    = (bf16_t*)alloc((size_t)NHD * NTOT * 128 * 2);
  float*  fsr   = (float*)alloc((size_t)NTOT * 64 * 4);
  bf16_t* edata = (bf16_t*)alloc((size_t)ECNT * LMP * 64 * 2);
  bf16_t* hA    = (bf16_t*)alloc((size_t)ECNT * 512 * 2);
  bf16_t* hB    = (bf16_t*)alloc((size_t)ECNT * 512 * 2);
  float*  abuf  = (float*)alloc((size_t)ECNT * NHD * 4);
  float*  ea    = (float*)alloc((size_t)ECNT * NHD * 4);
  unsigned* amax= (unsigned*)alloc((size_t)NLC * NHD * 4);
  float*  den   = (float*)alloc((size_t)NLC * NHD * 4);
  float*  ft    = (float*)alloc((size_t)NLC * 512 * 4);
  float*  mbuf[2];
  mbuf[0] = (float*)alloc((size_t)BTGT * 512 * 4);
  mbuf[1] = (float*)alloc((size_t)BTGT * 512 * 4);
  float*  sbuf  = (float*)alloc(16 * 4);
  // packed weights (fragment-major bf16, FRAGE elems each)
  bf16_t* fcW0p = (bf16_t*)alloc((size_t)63 * 4 * FRAGE * 2);
  bf16_t* fcW1p = (bf16_t*)alloc((size_t)47 * 4 * FRAGE * 2);
  bf16_t* fcW2p = (bf16_t*)alloc((size_t)32 * 4 * FRAGE * 2);
  bf16_t* acWp  = (bf16_t*)alloc((size_t)8 * 16 * FRAGE * 2);
  bf16_t* acW2p = (bf16_t*)alloc((size_t)8 * 32 * FRAGE * 2);
  bf16_t* wihP[2], *whhP[2];
  wihP[0] = (bf16_t*)alloc((size_t)192 * FRAGE * 2);
  wihP[1] = (bf16_t*)alloc((size_t)192 * FRAGE * 2);
  whhP[0] = (bf16_t*)alloc((size_t)1536 * FRAGE * 2);
  whhP[1] = (bf16_t*)alloc((size_t)1536 * FRAGE * 2);
  bf16_t* attnWp = (bf16_t*)alloc((size_t)16 * FRAGE * 2);
  bf16_t* Vp     = (bf16_t*)alloc((size_t)(MPAD / 32) * 4 * FRAGE * 2);

  auto cdiv = [](long a, long b) { return (unsigned)((a + b - 1) / b); };

  hipMemsetAsync(fsr, 0, (size_t)NTOT * 64 * 4, stream);
  hipMemsetAsync(sbuf, 0, 16 * 4, stream);

  // 0) pack all weights once (tiny, L2-resident afterwards)
  k_packB<<<cdiv((long)63 * 4 * 32, 256), 256, 0, stream>>>(fcW0, 2000, 64, 63, 1, fcW0p);
  k_packB<<<cdiv((long)47 * 4 * 32, 256), 256, 0, stream>>>(fcW1, 1500, 64, 47, 1, fcW1p);
  k_packB<<<cdiv((long)32 * 4 * 32, 256), 256, 0, stream>>>(fcW2, 1000, 64, 32, 1, fcW2p);
  k_packB<<<cdiv((long)8 * 16 * 32, 256), 256, 0, stream>>>(acW, 64, 128, 2, 8, acWp);
  k_packB<<<cdiv((long)8 * 32 * 32, 256), 256, 0, stream>>>(acW2, 128, 128, 4, 8, acW2p);
  for (int mp = 0; mp < PMP; ++mp) {
    k_packBT<<<cdiv((long)192 * 32, 256), 256, 0, stream>>>(
        gwih + (size_t)mp * G3H * 64, 64, G3H, wihP[mp]);
    k_packBT<<<cdiv((long)1536 * 32, 256), 256, 0, stream>>>(
        gwhh + (size_t)mp * G3H * 512, 512, G3H, whhP[mp]);
  }

  // 1) per-type feature transforms -> tf, tf16
  k_feat<<<(N0C + 63) / 64, 128, 0, stream>>>(f0, fcW0p, fcb0, N0C, 2000, 0, tf, tf16);
  k_feat<<<(N1C + 63) / 64, 128, 0, stream>>>(f1, fcW1p, fcb1, N1C, 1500, N0C, tf, tf16);
  k_feat<<<(N2C + 63) / 64, 128, 0, stream>>>(f2, fcW2p, fcb2, N2C, 1000, N0C + N1C, tf, tf16);
  k_packV<<<cdiv((long)(MPAD / 32) * 4 * 32, 256), 256, 0, stream>>>(tf16, Vp);

  // 2-3) attention projections
  dim3 gq((NTOT + 63) / 64, NHD);
  k_q0<<<gq, 128, 0, stream>>>(emb, acWp, q0);
  k_q<<<gq, 128, 0, stream>>>(q0, acW2p, qb);

  // 4) fused HGNN_AC attention -> fsr
  dim3 ga((NTOT + 15) / 16, NHD);
  k_attn<<<ga, 128, 0, stream>>>(qb, q0, Vp, fsr);

  // 5) tf[N0:] += fsr[N0:]
  k_addfsr<<<cdiv((long)(NTOT - N0C) * 64, 256), 256, 0, stream>>>(tf, tf16, fsr);

  // 6-12) per-metapath pipeline
  for (int mp = 0; mp < PMP; ++mp) {
    k_gather<<<cdiv((long)ECNT * LMP * 8, 256), 256, 0, stream>>>(emi[mp], tf16, edata);

    hipMemsetAsync(amax, 0, (size_t)NLC * NHD * 4, stream);
    hipMemsetAsync(den, 0, (size_t)NLC * NHD * 4, stream);
    hipMemsetAsync(ft, 0, (size_t)NLC * 512 * 4, stream);

    const float* bih = gbih + (size_t)mp * G3H;
    const float* bhh = gbhh + (size_t)mp * G3H;
    dim3 gg(4, ECNT / 16);  // 8 col tiles per block x 3125 row tiles
    k_gru<<<gg, 128, 0, stream>>>(edata,       LMP * 64, nullptr, wihP[mp], whhP[mp], bih, bhh, hA, 1);
    k_gru<<<gg, 128, 0, stream>>>(edata + 64,  LMP * 64, hA,      wihP[mp], whhP[mp], bih, bhh, hB, 0);
    k_gru<<<gg, 128, 0, stream>>>(edata + 128, LMP * 64, hB,      wihP[mp], whhP[mp], bih, bhh, hA, 0);
    // hA holds eft [E, 512]

    k_pack_attnW<<<2, 256, 0, stream>>>(attnp + (size_t)mp * NHD * 64, attnWp);
    k_scoresW<<<cdiv((long)(ECNT / 16), 4), 128, 0, stream>>>(hA, attnWp, edst[mp], abuf, amax);
    k_expden<<<cdiv((long)ECNT * NHD, 256), 256, 0, stream>>>(abuf, amax, edst[mp], ea, den);
    k_agg<<<cdiv((long)ECNT * 512, 256), 256, 0, stream>>>(hA, ea, den, edst[mp], ft);
    k_gelu<<<cdiv((long)BTGT * 512, 256), 256, 0, stream>>>(ft, tidx[mp], mbuf[mp]);
    k_sem<<<BTGT, 128, 0, stream>>>(mbuf[mp], fc1W, fc1b, fc2w, sbuf + mp);
  }

  // 13) semantic softmax + final output
  k_beta<<<1, 32, 0, stream>>>(sbuf, sbuf + 2);
  k_final<<<BTGT, 128, 0, stream>>>(mbuf[0], mbuf[1], sbuf + 2, outW, outb, out);
}
